// UltraCompoundRecurrentFFN_69123203662008
// MI455X (gfx1250) — compile-verified, matrix operand order checked by reading, not run on hardware
//
#include <hip/hip_runtime.h>
#include <math.h>

// ---------------- problem constants ----------------
#define BB   4
#define TT   2048
#define HH   1024
#define SS   512
#define HALF 256
#define MM   (BB*TT)        // 8192 rows
#define N1P  6464           // stage-1 fused output cols (6403 padded to 64*101)
#define K2   2048           // stage-2 K = 512 + 512 + 1024
#define HO   1024           // output width

typedef __bf16 bf16;
typedef __attribute__((ext_vector_type(16))) __bf16 bf16x16;
typedef __attribute__((ext_vector_type(8)))  float  floatx8;
typedef int v4i __attribute__((ext_vector_type(4)));
typedef __attribute__((address_space(1))) v4i* v4i_gp;   // global int4 vector ptr
typedef __attribute__((address_space(3))) v4i* v4i_lp;   // LDS int4 vector ptr

union Frag16 { uint4 u[2]; bf16x16 v; };

// ---------------- CDNA5 async global->LDS helpers ----------------
#if __has_builtin(__builtin_amdgcn_global_load_async_to_lds_b128)
#define HAS_ASYNC_LDS 1
#else
#define HAS_ASYNC_LDS 0
#endif

__device__ __forceinline__ void async_g2l_b128(const bf16* g, bf16* l) {
#if HAS_ASYNC_LDS
    __builtin_amdgcn_global_load_async_to_lds_b128((v4i_gp)(void*)g, (v4i_lp)l, 0, 0);
#else
    *(uint4*)l = *(const uint4*)g;   // sync fallback
#endif
}

__device__ __forceinline__ void wait_async0() {
#if __has_builtin(__builtin_amdgcn_s_wait_asynccnt)
    __builtin_amdgcn_s_wait_asynccnt(0);
#elif HAS_ASYNC_LDS
    asm volatile("s_wait_asynccnt 0" ::: "memory");
#endif
}

// ---------------- device helpers ----------------
__device__ __forceinline__ float sigmoidf_(float v) { return 1.0f / (1.0f + __expf(-v)); }
__device__ __forceinline__ float siluf_(float v)    { return v * sigmoidf_(v); }

// Fused stage-1 activation, selected by global output column.
__device__ __forceinline__ float act_apply(float v, int col) {
    if (col < 512)        return tanhf(v);                 // rf
    else if (col < 1024)  return sigmoidf_(v);             // rg (sigmoid)
    else if (col < 1536)  return siluf_(v);                // rc (silu)
    else if (col < 2048)  return v;                        // qr raw
    else if (col < 2560)  return siluf_(v);                // sc (silu)
    else if (col < 3072)  return sigmoidf_(v);             // sg (sigmoid)
    else if (col < 3584)  return sigmoidf_(-v);            // damp = exp(-softplus(v))
    else if (col < 4096)  return v;                        // qs raw
    else if (col < 4352)  return 3.14159265358979f * tanhf(v); // omega
    else if (col < 5376)  return v;                        // up raw
    else if (col < 6400)  return siluf_(v);                // gate (silu)
    else                  return v;                        // mix logits / pad
}

// ---------------- conversion / packing kernels ----------------
__global__ __launch_bounds__(256) void cvt_f32_bf16_k(const float* __restrict__ in,
                                                      bf16* __restrict__ out, int n) {
    int i = blockIdx.x * 256 + threadIdx.x;
    if (i < n) out[i] = (bf16)in[i];
}

// Wt1[n][k] = concat-weight[k][n], n in [0,N1P), k in [0,H)
__global__ __launch_bounds__(256) void build_w1_k(
    const float* __restrict__ rf, const float* __restrict__ ri,
    const float* __restrict__ rv, const float* __restrict__ rq,
    const float* __restrict__ sw, const float* __restrict__ sg,
    const float* __restrict__ sd, const float* __restrict__ sq,
    const float* __restrict__ som, const float* __restrict__ up,
    const float* __restrict__ gate, const float* __restrict__ mix,
    bf16* __restrict__ wt1) {
    int idx = blockIdx.x * 256 + threadIdx.x;
    if (idx >= N1P * HH) return;
    int n = idx >> 10, k = idx & 1023;
    float v = 0.0f;
    if (n < 4096) {
        int gi = n >> 9, col = n & 511;
        const float* gp = gi == 0 ? rf : gi == 1 ? ri : gi == 2 ? rv : gi == 3 ? rq
                        : gi == 4 ? sw : gi == 5 ? sg : gi == 6 ? sd : sq;
        v = gp[(size_t)k * 512 + col];
    } else if (n < 4352) v = som[(size_t)k * 256 + (n - 4096)];
    else if (n < 5376)   v = up  [(size_t)k * 1024 + (n - 4352)];
    else if (n < 6400)   v = gate[(size_t)k * 1024 + (n - 5376)];
    else if (n < 6403)   v = mix [(size_t)k * 3    + (n - 6400)];
    wt1[idx] = (bf16)v;
}

// Wt2[n][k] = [W_ro; W_so; W_down][k][n], n in [0,HO), k in [0,K2)
__global__ __launch_bounds__(256) void build_w2_k(
    const float* __restrict__ ro, const float* __restrict__ so,
    const float* __restrict__ down, bf16* __restrict__ wt2) {
    int idx = blockIdx.x * 256 + threadIdx.x;
    if (idx >= HO * K2) return;
    int n = idx >> 11, k = idx & 2047;
    float v;
    if (k < 512)       v = ro  [(size_t)k * HH + n];
    else if (k < 1024) v = so  [(size_t)(k - 512) * HH + n];
    else               v = down[(size_t)(k - 1024) * HH + n];
    wt2[idx] = (bf16)v;
}

// ---------------- WMMA bf16 GEMM: C(MxN) = A(MxK) * Bt(NxK)^T ----------------
// Block: 256 thr (8 waves), 128x64 tile, KT=64, double-buffered async LDS staging.
#define KT    64
#define LDSA  72   // row stride (elements), pads banks; 144B rows stay 16B aligned

template <bool ACT>
__global__ __launch_bounds__(256) void gemm_bf16_k(
    const bf16* __restrict__ A, int lda,
    const bf16* __restrict__ Bt, int ldb,
    float* __restrict__ C, int ldc, int K) {
    __shared__ __align__(16) bf16 As[2][128 * LDSA];
    __shared__ __align__(16) bf16 Bs[2][64 * LDSA];

    const int tid  = threadIdx.x;
    const int lane = tid & 31;
    const int wave = tid >> 5;
    const int wm   = wave & 3;    // 4 waves along M (32 rows each -> 128)
    const int wn   = wave >> 2;   // 2 waves along N (32 cols each -> 64)
    const int mbase = blockIdx.x * 128;
    const int nbase = blockIdx.y * 64;
    const int half  = lane >> 4;  // ISA 16-bit operand layout: K 0..7/16..23 vs 8..15/24..31
    const int l16   = lane & 15;

    floatx8 acc[2][2];
    const floatx8 zero = {0.f, 0.f, 0.f, 0.f, 0.f, 0.f, 0.f, 0.f};
#pragma unroll
    for (int i = 0; i < 2; ++i)
#pragma unroll
        for (int j = 0; j < 2; ++j) acc[i][j] = zero;

    // stage loader: A tile 128xKT (1024 16B chunks), B tile 64xKT (512 chunks)
    auto stage_load = [&](int buf, int kt) {
#pragma unroll
        for (int i = 0; i < 4; ++i) {
            int c = tid + 256 * i;              // 0..1023
            int row = c >> 3, part = c & 7;
            async_g2l_b128(A + (size_t)(mbase + row) * lda + kt + part * 8,
                           &As[buf][row * LDSA + part * 8]);
        }
#pragma unroll
        for (int i = 0; i < 2; ++i) {
            int c = tid + 256 * i;              // 0..511
            int row = c >> 3, part = c & 7;
            async_g2l_b128(Bt + (size_t)(nbase + row) * ldb + kt + part * 8,
                           &Bs[buf][row * LDSA + part * 8]);
        }
    };

    stage_load(0, 0);

    int buf = 0;
    for (int kt = 0; kt < K; kt += KT, buf ^= 1) {
        wait_async0();       // this wave's async fills of `buf` are complete
        __syncthreads();     // ... and everyone else's too

        if (kt + KT < K) {
            stage_load(buf ^ 1, kt + KT);   // overlap next tile with compute
            __builtin_prefetch(A + (size_t)(mbase + (tid >> 1)) * lda + kt + 2 * KT, 0, 1);
            __builtin_prefetch(Bt + (size_t)(nbase + (tid >> 2)) * ldb + kt + 2 * KT, 0, 1);
        }

#pragma unroll
        for (int ks = 0; ks < KT; ks += 32) {
            Frag16 af[2], bfr[2];
#pragma unroll
            for (int mi = 0; mi < 2; ++mi) {
                const bf16* p = &As[buf][(wm * 32 + mi * 16 + l16) * LDSA + ks + 8 * half];
                af[mi].u[0] = *(const uint4*)p;         // K ks..ks+7  (or +8.. for half=1)
                af[mi].u[1] = *(const uint4*)(p + 16);  // K ks+16..   (or +24..)
            }
#pragma unroll
            for (int ni = 0; ni < 2; ++ni) {
                const bf16* p = &Bs[buf][(wn * 32 + ni * 16 + l16) * LDSA + ks + 8 * half];
                bfr[ni].u[0] = *(const uint4*)p;
                bfr[ni].u[1] = *(const uint4*)(p + 16);
            }
#pragma unroll
            for (int mi = 0; mi < 2; ++mi)
#pragma unroll
                for (int ni = 0; ni < 2; ++ni)
                    acc[mi][ni] = __builtin_amdgcn_wmma_f32_16x16x32_bf16(
                        false, af[mi].v, false, bfr[ni].v, (short)0, acc[mi][ni], false, false);
        }
    }

    // epilogue: C 16x16 layout — lane<16: N=lane,M=j ; lane>=16: N=lane-16,M=j+8
#pragma unroll
    for (int mi = 0; mi < 2; ++mi) {
#pragma unroll
        for (int ni = 0; ni < 2; ++ni) {
            int n  = nbase + wn * 32 + ni * 16 + l16;
            int m0 = mbase + wm * 32 + mi * 16 + 8 * half;
#pragma unroll
            for (int j = 0; j < 8; ++j) {
                float v = acc[mi][ni][j];
                if (ACT) v = act_apply(v, n);
                C[(size_t)(m0 + j) * ldc + n] = v;
            }
        }
    }
}

// ---------------- mix softmax ----------------
__global__ __launch_bounds__(256) void mix_softmax_k(const float* __restrict__ P1,
                                                     float* __restrict__ mwb) {
    int m = blockIdx.x * 256 + threadIdx.x;
    if (m >= MM) return;
    const float* p = P1 + (size_t)m * N1P + 6400;
    float l0 = p[0], l1 = p[1], l2 = p[2];
    float mx = fmaxf(l0, fmaxf(l1, l2));
    float e0 = __expf(l0 - mx), e1 = __expf(l1 - mx), e2 = __expf(l2 - mx);
    float inv = 1.0f / (e0 + e1 + e2);
    mwb[m * 4 + 0] = e0 * inv;
    mwb[m * 4 + 1] = e1 * inv;
    mwb[m * 4 + 2] = e2 * inv;
}

// ---------------- sequential scan (fused stage-2 A-operand emit) ----------------
__global__ __launch_bounds__(256) void scan_k(const float* __restrict__ P1,
                                              const float* __restrict__ mwb,
                                              const float* __restrict__ read_init,
                                              const float* __restrict__ stable_init,
                                              bf16* __restrict__ A2) {
    const int b = blockIdx.x;        // batch
    const int i = threadIdx.x;       // pair index 0..255
    float rs0 = read_init[i], rs1 = read_init[i + HALF];
    float se  = stable_init[i], so = stable_init[i + HALF];

#pragma unroll 1
    for (int t = 0; t < TT; ++t) {
        const size_t row = (size_t)(b * TT + t);
        const float* Pm = P1 + row * N1P;
        if (t + 1 < TT) __builtin_prefetch(Pm + N1P + i, 0, 1);

        float rf0 = Pm[i],          rf1 = Pm[i + HALF];
        float rg0 = Pm[512 + i],    rg1 = Pm[512 + i + HALF];
        float rc0 = Pm[1024 + i],   rc1 = Pm[1024 + i + HALF];
        float qr0 = Pm[1536 + i],   qr1 = Pm[1536 + i + HALF];
        float sc0 = Pm[2048 + i],   sc1 = Pm[2048 + i + HALF];
        float sg0 = Pm[2560 + i],   sg1 = Pm[2560 + i + HALF];
        float dp0 = Pm[3072 + i],   dp1 = Pm[3072 + i + HALF];
        float qs0 = Pm[3584 + i],   qs1 = Pm[3584 + i + HALF];
        float om  = Pm[4096 + i];
        float mw0 = mwb[row * 4 + 0];
        float mw1 = mwb[row * 4 + 1];

        // read state
        rs0 = rf0 * rs0 + rg0 * rc0;
        rs1 = rf1 * rs1 + rg1 * rc1;
        // stable state: rotate pair (i, i+HALF) by omega, damp, inject
        float cw = __cosf(om), sw = __sinf(om);
        float ne = dp0 * (cw * se - sw * so) + sg0 * sc0;
        float no = dp1 * (sw * se + cw * so) + sg1 * sc1;
        se = ne; so = no;

        bf16* a2 = A2 + row * K2;
        a2[i]              = (bf16)(siluf_(qr0 * rs0) * mw0);
        a2[i + HALF]       = (bf16)(siluf_(qr1 * rs1) * mw0);
        a2[512 + i]        = (bf16)(siluf_(qs0 * se) * mw1);
        a2[512 + i + HALF] = (bf16)(siluf_(qs1 * so) * mw1);
    }
}

// ---------------- local branch: A2[:,1024:2048] = up * silu(gate) * mw2 ----------------
__global__ __launch_bounds__(256) void local_k(const float* __restrict__ P1,
                                               const float* __restrict__ mwb,
                                               bf16* __restrict__ A2) {
    int idx = blockIdx.x * 256 + threadIdx.x;
    if (idx >= MM * 1024) return;
    int m = idx >> 10, j = idx & 1023;
    const float* p = P1 + (size_t)m * N1P;
    float v = p[4352 + j] * p[5376 + j] * mwb[m * 4 + 2];
    A2[(size_t)m * K2 + 1024 + j] = (bf16)v;
}

// ---------------- host launch ----------------
extern "C" void kernel_launch(void* const* d_in, const int* in_sizes, int n_in,
                              void* d_out, int out_size, void* d_ws, size_t ws_size,
                              hipStream_t stream) {
    const float* x      = (const float*)d_in[0];
    const float* W_rf   = (const float*)d_in[1];
    const float* W_ri   = (const float*)d_in[2];
    const float* W_rv   = (const float*)d_in[3];
    const float* W_rq   = (const float*)d_in[4];
    const float* W_ro   = (const float*)d_in[5];
    const float* W_sw   = (const float*)d_in[6];
    const float* W_sg   = (const float*)d_in[7];
    const float* W_som  = (const float*)d_in[8];
    const float* W_sd   = (const float*)d_in[9];
    const float* W_sq   = (const float*)d_in[10];
    const float* W_so   = (const float*)d_in[11];
    const float* W_up   = (const float*)d_in[12];
    const float* W_gate = (const float*)d_in[13];
    const float* W_down = (const float*)d_in[14];
    const float* W_mix  = (const float*)d_in[15];
    const float* r_init = (const float*)d_in[16];
    const float* s_init = (const float*)d_in[17];
    float* out = (float*)d_out;

    char* ws = (char*)d_ws;
    bf16*  xb  = (bf16*)ws;  ws += (size_t)MM * HH * 2;
    bf16*  wt1 = (bf16*)ws;  ws += (size_t)N1P * HH * 2;
    bf16*  wt2 = (bf16*)ws;  ws += (size_t)HO * K2 * 2;
    float* P1  = (float*)ws; ws += (size_t)MM * N1P * 4;
    bf16*  A2  = (bf16*)ws;  ws += (size_t)MM * K2 * 2;
    float* mwb = (float*)ws;

    // 1) precision conversion + weight packing
    cvt_f32_bf16_k<<<(MM * HH + 255) / 256, 256, 0, stream>>>(x, xb, MM * HH);
    build_w1_k<<<(N1P * HH + 255) / 256, 256, 0, stream>>>(
        W_rf, W_ri, W_rv, W_rq, W_sw, W_sg, W_sd, W_sq, W_som, W_up, W_gate, W_mix, wt1);
    build_w2_k<<<(HO * K2 + 255) / 256, 256, 0, stream>>>(W_ro, W_so, W_down, wt2);

    // 2) stage-1 fused projection GEMM (M=8192, N=6464, K=1024) + activations
    gemm_bf16_k<true><<<dim3(MM / 128, N1P / 64), 256, 0, stream>>>(
        xb, HH, wt1, HH, P1, N1P, HH);

    // 3) mix softmax
    mix_softmax_k<<<(MM + 255) / 256, 256, 0, stream>>>(P1, mwb);

    // 4) sequential scan, emitting stage-2 A operand cols [0,1024)
    scan_k<<<BB, HALF, 0, stream>>>(P1, mwb, r_init, s_init, A2);

    // 5) local branch -> A2 cols [1024,2048)
    local_k<<<(MM * 1024 + 255) / 256, 256, 0, stream>>>(P1, mwb, A2);

    // 6) stage-2 output GEMM (M=8192, N=1024, K=2048) -> d_out
    gemm_bf16_k<false><<<dim3(MM / 128, HO / 64), 256, 0, stream>>>(
        A2, K2, wt2, K2, out, HO, K2);
}
